// GAT_61744449848088
// MI455X (gfx1250) — compile-verified, matrix operand order checked by reading, not run on hardware
//
#include <hip/hip_runtime.h>
#include <hip/hip_bf16.h>
#include <float.h>

#define N_NODES 50000
#define F_DIM 96
#define E_EDGES 800000
#define OUT_STRIDE 384   // F_DIM * (DEPTH+1)
#define ALPHA 0.2f

typedef __attribute__((ext_vector_type(16))) _Float16 v16h;
typedef __attribute__((ext_vector_type(8)))  float    v8f;

// ---- ordered-uint encoding for float atomic max (handles negatives) ----
__device__ __forceinline__ unsigned f2ord(float x) {
  unsigned u = __float_as_uint(x);
  return (u >> 31) ? ~u : (u | 0x80000000u);
}
__device__ __forceinline__ float ord2f(unsigned e) {
  unsigned u = (e >> 31) ? (e ^ 0x80000000u) : ~e;
  return __uint_as_float(u);
}

// ---- pack two aligned 8-float runs into a v16h WMMA fragment (scaled) ----
__device__ __forceinline__ v16h pack16(float4 x0, float4 x1, float4 y0, float4 y1,
                                       float s) {
  v16h v;
  v[0]  = (_Float16)(x0.x * s); v[1]  = (_Float16)(x0.y * s);
  v[2]  = (_Float16)(x0.z * s); v[3]  = (_Float16)(x0.w * s);
  v[4]  = (_Float16)(x1.x * s); v[5]  = (_Float16)(x1.y * s);
  v[6]  = (_Float16)(x1.z * s); v[7]  = (_Float16)(x1.w * s);
  v[8]  = (_Float16)(y0.x * s); v[9]  = (_Float16)(y0.y * s);
  v[10] = (_Float16)(y0.z * s); v[11] = (_Float16)(y0.w * s);
  v[12] = (_Float16)(y1.x * s); v[13] = (_Float16)(y1.y * s);
  v[14] = (_Float16)(y1.z * s); v[15] = (_Float16)(y1.w * s);
  return v;
}

// ---- zero a 96-wide column slice of the N x 384 output ----
__global__ void zero_cols_k(float* __restrict__ out, int base) {
  int g = blockIdx.x * blockDim.x + threadIdx.x;
  if (g >= N_NODES * F_DIM) return;
  int i = g / F_DIM, f = g - i * F_DIM;
  out[(size_t)i * OUT_STRIDE + base + f] = 0.f;
}

// ---- initial features: scatter-add adj_self_val * node_f via indices ----
__global__ void scatter_init_k(const float* __restrict__ nf,
                               const int* __restrict__ sidx,
                               const float* __restrict__ val,
                               float* __restrict__ out) {
  int g = blockIdx.x * blockDim.x + threadIdx.x;
  if (g >= N_NODES * F_DIM) return;
  int i = g / F_DIM, f = g - i * F_DIM;
  int r = sidx[2 * i], c = sidx[2 * i + 1];
  atomicAdd(&out[(size_t)r * OUT_STRIDE + f], val[i] * nf[(size_t)c * F_DIM + f]);
}

// ---- in-place relu on a column slice ----
__global__ void relu_cols_k(float* __restrict__ out, int base) {
  int g = blockIdx.x * blockDim.x + threadIdx.x;
  if (g >= N_NODES * F_DIM) return;
  int i = g / F_DIM, f = g - i * F_DIM;
  size_t p = (size_t)i * OUT_STRIDE + base + f;
  out[p] = fmaxf(out[p], 0.f);
}

// ---- attention coefficients via WMMA: [feats(N x 96)] x [W(96 x 4)] ----
// W columns: 0 = a_self[h0], 1 = a_neigh[h0], 2 = a_self[h1], 3 = a_neigh[h1]
// One wave computes a 16-row tile with 3 accumulating v_wmma_f32_16x16x32_f16.
// Branch-free fragment assembly: lanes n>=4 read a valid dummy pointer and
// multiply by 0, so no EXEC divergence and loads vectorize to b128.
__global__ void coeff_wmma_k(const float* __restrict__ feats,  // stride OUT_STRIDE
                             const float* __restrict__ a_self,
                             const float* __restrict__ a_neigh,
                             float* __restrict__ ecoef) {      // 4 arrays of N
  int lane = threadIdx.x & 31;
  int wave = threadIdx.x >> 5;
  int tile = blockIdx.x * (blockDim.x >> 5) + wave;
  if (tile * 16 >= N_NODES) return;          // wave-uniform exit (EXEC all-1s inside)

  int kb = (lane >> 4) << 3;                 // 0 or 8: K sub-block per lane half
  int r  = tile * 16 + (lane & 15);          // A-matrix row for this lane
  const float* frow = feats + (size_t)r * OUT_STRIDE;

  int n = lane & 15;                          // B-matrix column for this lane
  const float* wptr = (n == 1) ? a_neigh
                    : (n == 2) ? a_self + F_DIM
                    : (n == 3) ? a_neigh + F_DIM
                    : a_self;                 // n==0 and n>=4 (masked) read a_self
  float bscale = (n < 4) ? 1.f : 0.f;

  // B fragments are tile-invariant: build all 3 K-chunks up front.
  v16h bfrag[3];
  #pragma unroll
  for (int c = 0; c < 3; ++c) {
    const float4* w0 = (const float4*)(wptr + c * 32 + kb);
    const float4* w1 = (const float4*)(wptr + c * 32 + kb + 16);
    bfrag[c] = pack16(w0[0], w0[1], w1[0], w1[1], bscale);
  }

  v8f acc = {};
  #pragma unroll
  for (int c = 0; c < 3; ++c) {
    const float4* f0 = (const float4*)(frow + c * 32 + kb);
    const float4* f1 = (const float4*)(frow + c * 32 + kb + 16);
    v16h a = pack16(f0[0], f0[1], f1[0], f1[1], 1.f);
    acc = __builtin_amdgcn_wmma_f32_16x16x32_f16(false, a, false, bfrag[c],
                                                 (short)0, acc, false, false);
  }

  if (n < 4) {
    int rbase = tile * 16 + ((lane < 16) ? 0 : 8);  // C/D layout: VGPR v -> row v or v+8
    #pragma unroll
    for (int v = 0; v < 8; ++v)
      ecoef[(size_t)n * N_NODES + rbase + v] = acc[v];
  }
}

// ---- init per-node per-head max (ordered-uint -FLT_MAX) and denom (0) ----
__global__ void init_md_k(unsigned* __restrict__ maxu, float* __restrict__ denom) {
  int i = blockIdx.x * blockDim.x + threadIdx.x;
  if (i >= 2 * N_NODES) return;
  maxu[i]  = f2ord(-FLT_MAX);
  denom[i] = 0.f;
}

// ---- edge scores (leaky relu) + segment max ----
__global__ void score_max_k(const int* __restrict__ eidx,
                            const float* __restrict__ ecoef,
                            float* __restrict__ scores,
                            unsigned* __restrict__ maxu) {
  int e = blockIdx.x * blockDim.x + threadIdx.x;
  if (e >= E_EDGES) return;
  int r = eidx[2 * e], c = eidx[2 * e + 1];
  #pragma unroll
  for (int h = 0; h < 2; ++h) {
    float x = ecoef[(size_t)(2 * h) * N_NODES + r] +
              ecoef[(size_t)(2 * h + 1) * N_NODES + c];
    float s = x > 0.f ? x : ALPHA * x;
    scores[(size_t)h * E_EDGES + e] = s;
    atomicMax(&maxu[h * N_NODES + r], f2ord(s));
  }
}

// ---- exp(score - max) + segment sum ----
__global__ void exp_sum_k(const int* __restrict__ eidx,
                          float* __restrict__ scores,
                          const unsigned* __restrict__ maxu,
                          float* __restrict__ denom) {
  int e = blockIdx.x * blockDim.x + threadIdx.x;
  if (e >= E_EDGES) return;
  int r = eidx[2 * e];
  #pragma unroll
  for (int h = 0; h < 2; ++h) {
    float m  = ord2f(maxu[h * N_NODES + r]);
    float ex = __expf(scores[(size_t)h * E_EDGES + e] - m);
    scores[(size_t)h * E_EDGES + e] = ex;
    atomicAdd(&denom[h * N_NODES + r], ex);
  }
}

// ---- aggregation: one warp per edge, heads fused into one edge weight ----
__global__ void aggregate_k(const int* __restrict__ eidx,
                            const float* __restrict__ scores,
                            const float* __restrict__ denom,
                            const float* __restrict__ feats_cur,  // stride OUT_STRIDE
                            float* __restrict__ out_next) {       // stride OUT_STRIDE
  int lane = threadIdx.x & 31;
  int e = (blockIdx.x * blockDim.x + threadIdx.x) >> 5;
  if (e >= E_EDGES) return;
  int r = eidx[2 * e], c = eidx[2 * e + 1];
  float w = 0.5f * (scores[e]           / denom[r] +
                    scores[E_EDGES + e] / denom[N_NODES + r]);
  const float* src = feats_cur + (size_t)c * OUT_STRIDE;
  float*       dst = out_next  + (size_t)r * OUT_STRIDE;
  #pragma unroll
  for (int f = lane; f < F_DIM; f += 32)
    atomicAdd(&dst[f], w * src[f]);
}

extern "C" void kernel_launch(void* const* d_in, const int* in_sizes, int n_in,
                              void* d_out, int out_size, void* d_ws, size_t ws_size,
                              hipStream_t stream) {
  (void)in_sizes; (void)n_in; (void)out_size; (void)ws_size;
  const float* node_f  = (const float*)d_in[0];
  const int*   adj_idx = (const int*)  d_in[1];
  const int*   sidx    = (const int*)  d_in[2];
  const float* sval    = (const float*)d_in[3];
  const float* a_self  = (const float*)d_in[4];
  const float* a_neigh = (const float*)d_in[5];
  float* out = (float*)d_out;

  float*    ws     = (float*)d_ws;
  float*    ecoef  = ws;                               // 4*N
  unsigned* maxu   = (unsigned*)(ws + 4 * N_NODES);    // 2*N
  float*    denom  = ws + 6 * N_NODES;                 // 2*N
  float*    scores = ws + 8 * N_NODES;                 // 2*E

  const int NF      = N_NODES * F_DIM;
  const int nf_blk  = (NF + 255) / 256;
  const int e_blk   = (E_EDGES + 255) / 256;
  const int md_blk  = (2 * N_NODES + 255) / 256;
  const int agg_blk = (E_EDGES * 32 + 255) / 256;
  const int tiles   = (N_NODES + 15) / 16;             // 3125 (exact)
  const int wm_blk  = (tiles + 7) / 8;                 // 8 waves per 256-thread block

  // stage 0: initial features (zero -> scatter-add -> relu) into columns [0,96)
  zero_cols_k   <<<nf_blk, 256, 0, stream>>>(out, 0);
  scatter_init_k<<<nf_blk, 256, 0, stream>>>(node_f, sidx, sval, out);
  relu_cols_k   <<<nf_blk, 256, 0, stream>>>(out, 0);

  for (int L = 0; L < 3; ++L) {
    const float* fcur  = out + L * F_DIM;
    float*       fnext = out + (L + 1) * F_DIM;

    coeff_wmma_k<<<wm_blk, 256, 0, stream>>>(fcur, a_self, a_neigh, ecoef);
    init_md_k   <<<md_blk, 256, 0, stream>>>(maxu, denom);
    score_max_k <<<e_blk,  256, 0, stream>>>(adj_idx, ecoef, scores, maxu);
    exp_sum_k   <<<e_blk,  256, 0, stream>>>(adj_idx, scores, maxu, denom);
    zero_cols_k <<<nf_blk, 256, 0, stream>>>(out, (L + 1) * F_DIM);
    aggregate_k <<<agg_blk, 256, 0, stream>>>(adj_idx, scores, denom, fcur, fnext);
    relu_cols_k <<<nf_blk, 256, 0, stream>>>(out, (L + 1) * F_DIM);
  }
}
